// SelfAttention_2130303779057
// MI455X (gfx1250) — compile-verified, matrix operand order checked by reading, not run on hardware
//
#include <hip/hip_runtime.h>
#include <hip/hip_bf16.h>
#include <stdint.h>

// ---------------------------------------------------------------------------
// Self-attention without transpose:  S = x@x ; P = softmax(S) ; out = P@x
// B=8, N=K=2048, fp32 in/out.  Compute-bound (~275 GFLOP) -> bf16 WMMA path.
//
// GEMM staging uses CDNA5 GLOBAL_LOAD_ASYNC_TO_LDS_B128 (ASYNCcnt-tracked
// direct memory->LDS DMA): no staging VGPRs, no ds_store traffic, double-
// buffered so the async engine fills buffer^1 while WMMAs consume buffer.
//
// Workspace: [0,64M) xb bf16 | [64M,128M) xbT bf16 | [128M,256M) S fp32
//            [256M,320M) P bf16
// ---------------------------------------------------------------------------

typedef __attribute__((ext_vector_type(16))) __bf16     v16bf;
typedef __attribute__((ext_vector_type(8)))  float      v8f;
typedef __attribute__((ext_vector_type(4))) unsigned    u32x4;
typedef __attribute__((ext_vector_type(2))) unsigned    u32x2;
typedef __attribute__((ext_vector_type(4)))  float      f32x4;

constexpr int kN      = 2048;
constexpr int kBatch  = 8;
constexpr int kBM     = 128;   // block tile M
constexpr int kBN     = 128;   // block tile N
constexpr int kBK     = 32;    // K per LDS stage == K of one bf16 WMMA
constexpr int kLdsU4  = 5;     // 5 x uint4 = 80B per 32-bf16 row (64B + 16B pad)

union FragBF { u32x4 q[2]; v16bf v; };

__device__ __forceinline__ unsigned pack_bf16x2(float lo, float hi) {
  unsigned ul = __float_as_uint(lo), uh = __float_as_uint(hi);
  ul = (ul + 0x7FFFu + ((ul >> 16) & 1u)) >> 16;   // round-to-nearest-even
  uh = (uh + 0x7FFFu + ((uh >> 16) & 1u)) >> 16;
  return ul | (uh << 16);
}

// ---------------------------------------------------------------------------
// fp32 -> bf16 bulk convert (4 elements / thread), row-major copy
// ---------------------------------------------------------------------------
__global__ __launch_bounds__(256) void cvt_f32_bf16(const float* __restrict__ x,
                                                    __bf16* __restrict__ y) {
  size_t i = (size_t)blockIdx.x * blockDim.x + threadIdx.x;
  f32x4 f = ((const f32x4*)x)[i];
  u32x2 p; p.x = pack_bf16x2(f.x, f.y); p.y = pack_bf16x2(f.z, f.w);
  ((u32x2*)y)[i] = p;
}

// ---------------------------------------------------------------------------
// fp32 -> bf16 transposed copy: yT[b][n][k] = bf16(x[b][k][n]).
// 32x32 tile through padded LDS; all global traffic is >=8B coalesced.
// ---------------------------------------------------------------------------
__global__ __launch_bounds__(256) void cvt_transpose_bf16(const float* __restrict__ x,
                                                          __bf16* __restrict__ yT) {
  __shared__ unsigned short tile[32][36];   // [k][n], 72B row stride (8B aligned)
  const int t = threadIdx.x;
  const size_t bo = (size_t)blockIdx.z * kN * kN;
  const int r0 = blockIdx.y * 32;           // source row (k)
  const int c0 = blockIdx.x * 32;           // source col (n)

  {
    const int rr = t >> 3;                  // 0..31
    const int cc = (t & 7) * 4;             // 0,4,..,28
    f32x4 f = *(const f32x4*)(x + bo + (size_t)(r0 + rr) * kN + c0 + cc);
    u32x2 p; p.x = pack_bf16x2(f.x, f.y); p.y = pack_bf16x2(f.z, f.w);
    *(u32x2*)&tile[rr][cc] = p;
  }
  __syncthreads();
  {
    const int n  = t >> 3;                  // 0..31
    const int k4 = (t & 7) * 4;             // 0,4,..,28
    const unsigned a0 = tile[k4 + 0][n], a1 = tile[k4 + 1][n];
    const unsigned a2 = tile[k4 + 2][n], a3 = tile[k4 + 3][n];
    u32x2 o; o.x = a0 | (a1 << 16); o.y = a2 | (a3 << 16);
    *(u32x2*)(yT + bo + (size_t)(c0 + n) * kN + r0 + k4) = o;
  }
}

// ---------------------------------------------------------------------------
// Batched GEMM  C[b] = A[b](MxK) * B[b](KxN) with B given TRANSPOSED
// (Bt[n][k]).  bf16 WMMA, fp32 accumulate, async direct-to-LDS staging.
// grid = (N/128, N/128, batch), block = 256 (8 wave32: 4 along M x 2 along N)
// Each wave computes 32x64 = 2x4 tiles of 16x16.
// ---------------------------------------------------------------------------
__global__ __launch_bounds__(256) void gemm_wmma_bf16(const __bf16* __restrict__ A,
                                                      const __bf16* __restrict__ Bt,
                                                      float* __restrict__ C) {
  __shared__ u32x4 ldsA[2][kBM * kLdsU4];   // [row][K], 80B stride
  __shared__ u32x4 ldsB[2][kBN * kLdsU4];   // [col][K], 80B stride

  const int t     = threadIdx.x;
  const int lane  = t & 31;
  const int wid   = t >> 5;
  const int waveM = wid & 3;     // 0..3  -> 32-row slab
  const int waveN = wid >> 2;    // 0..1  -> 64-col slab
  const int kh    = lane >> 4;   // K-half select for 16-bit fragment layout
  const int rr    = lane & 15;   // row/col within 16x16 tile

  const int rowBase = blockIdx.y * kBM;
  const int colBase = blockIdx.x * kBN;
  const size_t bo   = (size_t)blockIdx.z * kN * kN;
  A  += bo; Bt += bo; C += bo;

  v8f acc[2][4];
  #pragma unroll
  for (int m = 0; m < 2; ++m)
    #pragma unroll
    for (int n = 0; n < 4; ++n)
      acc[m][n] = (v8f){0.f, 0.f, 0.f, 0.f, 0.f, 0.f, 0.f, 0.f};

  // Per-thread async-copy slice: 2 x b128 per matrix per K-step.
  // Tile rows/cols: rc = (t>>2) + i*64 ; 8 bf16 along K at (t&3)*8.
  const int ldRC  = t >> 2;            // 0..63
  const int ldK8  = (t & 3) * 8;       // 0,8,16,24
  const __bf16* aBase = A  + (size_t)(rowBase + ldRC) * kN + ldK8;
  const __bf16* bBase = Bt + (size_t)(colBase + ldRC) * kN + ldK8;

  unsigned ldsAoff[2][2], ldsBoff[2][2];
  #pragma unroll
  for (int b = 0; b < 2; ++b)
    #pragma unroll
    for (int i = 0; i < 2; ++i) {
      ldsAoff[b][i] = (unsigned)(size_t)&ldsA[b][0] + (ldRC + i * 64) * 80 + (t & 3) * 16;
      ldsBoff[b][i] = (unsigned)(size_t)&ldsB[b][0] + (ldRC + i * 64) * 80 + (t & 3) * 16;
    }

  // Issue this thread's share of one K-step stage: memory -> LDS, no VGPR data.
  auto issueStage = [&](int buf, int tkk) {
    const size_t kOff = (size_t)tkk * kBK;
    #pragma unroll
    for (int i = 0; i < 2; ++i) {
      const __bf16* ga = aBase + (size_t)i * 64 * kN + kOff;
      const __bf16* gb = bBase + (size_t)i * 64 * kN + kOff;
      asm volatile("global_load_async_to_lds_b128 %0, %1, off"
                   :: "v"(ldsAoff[buf][i]), "v"(ga) : "memory");
      asm volatile("global_load_async_to_lds_b128 %0, %1, off"
                   :: "v"(ldsBoff[buf][i]), "v"(gb) : "memory");
    }
  };
  auto waitAsync = [&]() {
    asm volatile("s_wait_asynccnt 0x0" ::: "memory");
  };

  // One K-step of WMMAs: A fragments pinned, B fragments streamed.
  auto mmaStep = [&](const u32x4* fA, const u32x4* fB) {
    FragBF fa0, fa1;
    const int ab = (waveM * 32 + rr) * kLdsU4 + kh;
    fa0.q[0] = fA[ab];          fa0.q[1] = fA[ab + 2];
    fa1.q[0] = fA[ab + 16 * kLdsU4]; fa1.q[1] = fA[ab + 16 * kLdsU4 + 2];
    #pragma unroll
    for (int n = 0; n < 4; ++n) {
      FragBF fb;
      const int bb = (waveN * 64 + n * 16 + rr) * kLdsU4 + kh;
      fb.q[0] = fB[bb]; fb.q[1] = fB[bb + 2];
      acc[0][n] = __builtin_amdgcn_wmma_f32_16x16x32_bf16(
          false, fa0.v, false, fb.v, (short)0, acc[0][n], false, false);
      acc[1][n] = __builtin_amdgcn_wmma_f32_16x16x32_bf16(
          false, fa1.v, false, fb.v, (short)0, acc[1][n], false, false);
    }
  };

  // Pipeline: async engine fills buf^1 while WMMAs consume buf.
  const int T = kN / kBK;   // 64 K-steps (even)
  issueStage(0, 0);
  waitAsync();
  __syncthreads();
  for (int tk = 0; tk < T; tk += 2) {
    // half A: compute buf0, prefetch buf1 (tk+1 < T always, T even)
    issueStage(1, tk + 1);
    mmaStep(ldsA[0], ldsB[0]);
    waitAsync();
    __syncthreads();
    // half B: compute buf1, prefetch buf0
    if (tk + 2 < T) issueStage(0, tk + 2);
    mmaStep(ldsA[1], ldsB[1]);
    if (tk + 2 < T) { waitAsync(); __syncthreads(); }
  }

  // C/D layout: VGPR i -> M = i (lanes 0-15) / i+8 (lanes 16-31), N = lane&15
  #pragma unroll
  for (int m = 0; m < 2; ++m)
    #pragma unroll
    for (int n = 0; n < 4; ++n)
      #pragma unroll
      for (int i = 0; i < 8; ++i) {
        const int row = rowBase + waveM * 32 + m * 16 + i + kh * 8;
        const int col = colBase + waveN * 64 + n * 16 + rr;
        C[(size_t)row * kN + col] = acc[m][n][i];
      }
}

// ---------------------------------------------------------------------------
// Row softmax, fp32 scores -> bf16 probabilities.
// One 256-thread block per 2048-element row.
// ---------------------------------------------------------------------------
__global__ __launch_bounds__(256) void softmax_bf16(const float* __restrict__ S,
                                                    __bf16* __restrict__ P) {
  __shared__ float red[8];
  const int t = threadIdx.x;
  const f32x4* row = (const f32x4*)(S + (size_t)blockIdx.x * kN);
  f32x4 a = row[t], b = row[t + 256];

  float m = fmaxf(fmaxf(fmaxf(a.x, a.y), fmaxf(a.z, a.w)),
                  fmaxf(fmaxf(b.x, b.y), fmaxf(b.z, b.w)));
  #pragma unroll
  for (int off = 16; off; off >>= 1) m = fmaxf(m, __shfl_xor(m, off));
  if ((t & 31) == 0) red[t >> 5] = m;
  __syncthreads();
  m = red[0];
  #pragma unroll
  for (int i = 1; i < 8; ++i) m = fmaxf(m, red[i]);

  a.x = __expf(a.x - m); a.y = __expf(a.y - m);
  a.z = __expf(a.z - m); a.w = __expf(a.w - m);
  b.x = __expf(b.x - m); b.y = __expf(b.y - m);
  b.z = __expf(b.z - m); b.w = __expf(b.w - m);
  float s = a.x + a.y + a.z + a.w + b.x + b.y + b.z + b.w;
  #pragma unroll
  for (int off = 16; off; off >>= 1) s += __shfl_xor(s, off);
  __syncthreads();                 // red[] reuse guard
  if ((t & 31) == 0) red[t >> 5] = s;
  __syncthreads();
  s = red[0] + red[1] + red[2] + red[3] + red[4] + red[5] + red[6] + red[7];
  const float inv = 1.0f / s;

  u32x2* prow = (u32x2*)(P + (size_t)blockIdx.x * kN);
  u32x2 o0, o1;
  o0.x = pack_bf16x2(a.x * inv, a.y * inv);
  o0.y = pack_bf16x2(a.z * inv, a.w * inv);
  o1.x = pack_bf16x2(b.x * inv, b.y * inv);
  o1.y = pack_bf16x2(b.z * inv, b.w * inv);
  prow[t] = o0; prow[t + 256] = o1;
}

// ---------------------------------------------------------------------------
extern "C" void kernel_launch(void* const* d_in, const int* in_sizes, int n_in,
                              void* d_out, int out_size, void* d_ws, size_t ws_size,
                              hipStream_t stream) {
  (void)in_sizes; (void)n_in; (void)out_size; (void)ws_size;
  const float* x   = (const float*)d_in[0];
  float*       out = (float*)d_out;
  char*        ws  = (char*)d_ws;

  const size_t total = (size_t)kBatch * kN * kN;           // 33.5M elements
  __bf16* xb  = (__bf16*)ws;                               // 64 MB, row-major
  __bf16* xbT = (__bf16*)(ws + total * 2);                 // 64 MB, transposed
  float*  S   = (float*)(ws + total * 4);                  // 128 MB fp32 scores
  __bf16* P   = (__bf16*)(ws + total * 8);                 // 64 MB bf16 probs

  // 1) x -> bf16 (row-major, A of GEMM #1)
  cvt_f32_bf16<<<dim3((unsigned)(total / 1024)), 256, 0, stream>>>(x, xb);
  // 2) x -> bf16 transposed (B of both GEMMs)
  cvt_transpose_bf16<<<dim3(kN / 32, kN / 32, kBatch), 256, 0, stream>>>(x, xbT);

  const dim3 g(kN / kBN, kN / kBM, kBatch);
  // 3) S = xb @ x   (bf16 WMMA, fp32 out)
  gemm_wmma_bf16<<<g, 256, 0, stream>>>(xb, xbT, S);
  // 4) P = softmax(S) as bf16
  softmax_bf16<<<dim3((unsigned)(kBatch * kN)), 256, 0, stream>>>(S, P);
  // 5) out = P @ x
  gemm_wmma_bf16<<<g, 256, 0, stream>>>(P, xbT, out);
}
